// ReverseMapping_41850161332643
// MI455X (gfx1250) — matrix operand order repaired and checked by reading, not compiled, not run on hardware
//
#include <hip/hip_runtime.h>
#include <hip/hip_bf16.h>

typedef _Float16 v8h  __attribute__((ext_vector_type(8)));
typedef _Float16 v16h __attribute__((ext_vector_type(16)));
typedef float    v8f  __attribute__((ext_vector_type(8)));
typedef int      v4i  __attribute__((vector_size(16)));   // matches builtin param

#define ZD   8
#define YD   112
#define XD   112
#define HF   28
#define WF   60
#define NCAM 6
#define CCH  128
#define CBEV 1024      // C*Z
#define PADW 114       // 112 + 2 halo
#define NKCHUNK 288    // 9 taps * 32 chunks of K=32

union ABfrag { v16h v; v8h h[2]; };

// ---- async global->LDS staging (gfx1250 ASYNCcnt path), guarded ----
#define AS1 __attribute__((address_space(1)))
#define AS3 __attribute__((address_space(3)))
#if defined(__has_builtin)
#if __has_builtin(__builtin_amdgcn_global_load_async_to_lds_b128)
#define USE_ASYNC 1
#endif
#endif
#ifndef USE_ASYNC
#define USE_ASYNC 0
#endif

__device__ __forceinline__ void wait_async_done() {
#if USE_ASYNC
#if defined(__has_builtin) && __has_builtin(__builtin_amdgcn_s_wait_asynccnt)
  __builtin_amdgcn_s_wait_asynccnt(0);
#else
  asm volatile("s_wait_asynccnt 0" ::: "memory");
#endif
#endif
}

// stage 32 B: global src -> LDS dst (per-thread slice of an 8KB chunk)
__device__ __forceinline__ void stage_async(const _Float16* src, _Float16* d) {
#if USE_ASYNC
  __builtin_amdgcn_global_load_async_to_lds_b128((AS1 v4i*)src,       (AS3 v4i*)d,       0, 0);
  __builtin_amdgcn_global_load_async_to_lds_b128((AS1 v4i*)(src + 8), (AS3 v4i*)(d + 8), 0, 0);
#else
  v8h a0 = ((const v8h*)src)[0];
  v8h a1 = ((const v8h*)src)[1];
  ((v8h*)d)[0] = a0;
  ((v8h*)d)[1] = a1;
#endif
}

// ---------------------------------------------------------------------------
// K0: per-camera projection matrices: P = K4 @ ext_inv @ inv(view)
// ---------------------------------------------------------------------------
__global__ void setup_mats_kernel(const float* __restrict__ intr,
                                  const float* __restrict__ extr,
                                  const float* __restrict__ view,
                                  float* __restrict__ mats) {
  if (threadIdx.x != 0 || blockIdx.x != 0) return;
  float a[4][8];
  for (int i = 0; i < 4; i++)
    for (int j = 0; j < 4; j++) { a[i][j] = view[i*4+j]; a[i][4+j] = (i==j)?1.f:0.f; }
  for (int c = 0; c < 4; c++) {
    int p = c; float mx = fabsf(a[c][c]);
    for (int r = c+1; r < 4; r++) { float v = fabsf(a[r][c]); if (v > mx) { mx = v; p = r; } }
    if (p != c) for (int j = 0; j < 8; j++) { float t = a[c][j]; a[c][j] = a[p][j]; a[p][j] = t; }
    float inv = 1.f / a[c][c];
    for (int j = 0; j < 8; j++) a[c][j] *= inv;
    for (int r = 0; r < 4; r++) if (r != c) {
      float f = a[r][c];
      for (int j = 0; j < 8; j++) a[r][j] -= f * a[c][j];
    }
  }
  float iv[4][4];
  for (int i = 0; i < 4; i++) for (int j = 0; j < 4; j++) iv[i][j] = a[i][4+j];

  for (int cam = 0; cam < NCAM; cam++) {
    const float* K = intr + cam*9;
    const float* E = extr + cam*16;
    float ks[3][3];
    for (int j = 0; j < 3; j++) { ks[0][j] = K[j]*0.125f; ks[1][j] = K[3+j]*0.125f; ks[2][j] = K[6+j]; }
    float R[3][3], t[3], Ei[4][4];
    for (int i = 0; i < 3; i++) { for (int j = 0; j < 3; j++) R[i][j] = E[i*4+j]; t[i] = E[i*4+3]; }
    for (int i = 0; i < 3; i++) for (int j = 0; j < 3; j++) Ei[i][j] = R[j][i];
    for (int i = 0; i < 3; i++) { float s = 0.f; for (int j = 0; j < 3; j++) s += R[j][i]*t[j]; Ei[i][3] = -s; }
    Ei[3][0]=0.f; Ei[3][1]=0.f; Ei[3][2]=0.f; Ei[3][3]=1.f;
    float T[4][4];
    for (int i = 0; i < 4; i++) for (int j = 0; j < 4; j++) {
      float s = 0.f; for (int k = 0; k < 4; k++) s += Ei[i][k]*iv[k][j]; T[i][j] = s;
    }
    float* o = mats + cam*16;
    for (int i = 0; i < 3; i++) for (int j = 0; j < 4; j++) {
      float s = 0.f; for (int k = 0; k < 3; k++) s += ks[i][k]*T[k][j]; o[i*4+j] = s;
    }
    for (int j = 0; j < 4; j++) o[12+j] = T[2][j];
  }
}

// ---------------------------------------------------------------------------
// K1: transpose feats (cam, C, H, W) -> (cam, H, W, C)
// ---------------------------------------------------------------------------
__global__ void transpose_feats_kernel(const float* __restrict__ in, float* __restrict__ out) {
  int idx = blockIdx.x * blockDim.x + threadIdx.x;
  const int total = NCAM*CCH*HF*WF;
  if (idx >= total) return;
  int c    = idx & (CCH-1);
  int rest = idx >> 7;
  int hw   = rest % (HF*WF);
  int cam  = rest / (HF*WF);
  out[idx] = in[(cam*CCH + c)*(HF*WF) + hw];
}

__global__ void zero_u32_kernel(unsigned* __restrict__ p, int n) {
  int i = blockIdx.x * blockDim.x + threadIdx.x;
  if (i < n) p[i] = 0u;
}

// ---------------------------------------------------------------------------
// K2: splat — one block per voxel, 128 threads = channels
// ---------------------------------------------------------------------------
__global__ void splat_kernel(const float* __restrict__ featsT,
                             const float* __restrict__ mats,
                             _Float16* __restrict__ bev) {
  int vox = blockIdx.x;
  int c   = threadIdx.x;
  int x =  vox % XD;
  int y = (vox / XD) % YD;
  int z =  vox / (XD*YD);
  float fx = (float)x, fy = (float)y, fz = (float)z;

  float sum = 0.f, cnt = 0.f;
  for (int cam = 0; cam < NCAM; cam++) {
    const float* M = mats + cam*16;
    float pxh = M[0]*fx + M[1]*fy + M[2] *fz + M[3];
    float pyh = M[4]*fx + M[5]*fy + M[6] *fz + M[7];
    float pz  = M[8]*fx + M[9]*fy + M[10]*fz + M[11];
    float zc  = M[12]*fx + M[13]*fy + M[14]*fz + M[15];
    float dz = fmaxf(pz, 1e-6f);
    float xp = pxh / dz, yp = pyh / dz;
    bool valid = (xp > -0.5f) && (xp < (float)WF - 0.5f) &&
                 (yp > -0.5f) && (yp < (float)HF - 0.5f) && (zc > 0.f);
    float x0 = floorf(xp), y0 = floorf(yp);
    float wx = xp - x0, wy = yp - y0;
    int ix0 = (int)x0, iy0 = (int)y0;
    const float* img = featsT + cam*(HF*WF*CCH);
    float acc = 0.f;
#define TAP(IX, IY, W) {                                                     \
      int ix = (IX), iy = (IY);                                              \
      bool inb = (ix >= 0) && (ix <= WF-1) && (iy >= 0) && (iy <= HF-1);     \
      int ixc = ix < 0 ? 0 : (ix > WF-1 ? WF-1 : ix);                        \
      int iyc = iy < 0 ? 0 : (iy > HF-1 ? HF-1 : iy);                        \
      acc += img[(iyc*WF + ixc)*CCH + c] * (inb ? (W) : 0.f); }
    TAP(ix0,   iy0,   (1.f-wx)*(1.f-wy));
    TAP(ix0+1, iy0,   wx*(1.f-wy));
    TAP(ix0,   iy0+1, (1.f-wx)*wy);
    TAP(ix0+1, iy0+1, wx*wy);
#undef TAP
    float val = valid ? acc : 0.f;
    float m   = (fabsf(val) > 0.f) ? 1.f : 0.f;
    sum += val;
    cnt += m;
  }
  float feat = sum / (1e-5f + cnt);
  bev[((y+1)*PADW + (x+1))*CBEV + c*ZD + z] = (_Float16)feat;
}

// ---------------------------------------------------------------------------
// K3: pack conv weights (oc, ci, 3, 3) f32 -> chunk-contiguous f16:
//     wt[k = t9*32+kc][oc][ci32]  (each chunk = 8 KB)
// ---------------------------------------------------------------------------
__global__ void pack_w_kernel(const float* __restrict__ w, _Float16* __restrict__ wt) {
  int o = blockIdx.x * blockDim.x + threadIdx.x;
  const int total = NKCHUNK*CCH*32;
  if (o >= total) return;
  int c32 = o & 31;
  int oc  = (o >> 5) & (CCH-1);
  int k   = o >> 12;        // chunk id 0..287
  int t9  = k >> 5;
  int kc  = k & 31;
  int ci  = kc*32 + c32;
  int ky  = t9 / 3, kx = t9 % 3;
  wt[o] = (_Float16)w[((oc*CBEV + ci)*3 + ky)*3 + kx];
}

// ---------------------------------------------------------------------------
// K4: implicit-GEMM 3x3 conv, V_WMMA_F32_16X16X32_F16, double-buffered async
// LDS weight staging, fused BN/ReLU. Loop-carried pointer swap + nounroll to
// keep accumulators pinned (no v_dual_mov shuffles / hazard nops), all
// addresses strictly incremental.
// ---------------------------------------------------------------------------
__global__ void __launch_bounds__(256, 1)
conv_gemm_kernel(const _Float16* __restrict__ bev,
                 const _Float16* __restrict__ wt,
                 const float* __restrict__ gamma, const float* __restrict__ beta,
                 const float* __restrict__ mean,  const float* __restrict__ var,
                 float* __restrict__ out) {
  __shared__ _Float16 ldsB[2][CCH * 32];   // 2 x 8 KB weight chunk [oc][k32]

  int tid  = threadIdx.x;
  int wave = tid >> 5;
  int lane = tid & 31;
  int tile = blockIdx.x * 4 + (wave >> 1);   // 0..783 spatial 16-tile
  int noff = (wave & 1) * 4;                 // oc-tile offset: 0 or 4
  int ty   = tile / 7;
  int tx0  = (tile % 7) * 16;
  int Ml   = lane & 15;
  int half = lane >> 4;

  v8f acc[4];
  v8f zero = {};
#pragma unroll
  for (int i = 0; i < 4; i++) acc[i] = zero;

  // per-buffer precomputed LDS addresses (swapped each K-step)
  _Float16* rd_cur = &ldsB[0][0] + (noff*16 + Ml)*32 + half*16;  // B-frag read base
  _Float16* rd_nxt = &ldsB[1][0] + (noff*16 + Ml)*32 + half*16;
  _Float16* st_cur = &ldsB[0][0] + tid*16;                       // stage dst base
  _Float16* st_nxt = &ldsB[1][0] + tid*16;

  // linear stage source: chunk k at wt + k*4096 halves
  const _Float16* wsrc = wt + tid*16;
  stage_async(wsrc, st_cur);
  wsrc += 4096;
  wait_async_done();
  __syncthreads();

  for (int t9 = 0; t9 < 9; t9++) {
    int ky = (t9 * 11) >> 5;          // t9 / 3 for t9 in [0,8]
    int kx = t9 - ky * 3;
    // A fragment pointer: advances 32 halves per K-step
    const _Float16* ap = bev + ((ty + ky)*PADW + (tx0 + Ml + kx))*CBEV + half*16;

#pragma clang loop unroll(disable)
    for (int kc = 0; kc < 32; kc++) {
      if (!(t9 == 8 && kc == 31)) {
        stage_async(wsrc, st_nxt);
        wsrc += 4096;
      }

      // A fragment (16x32 f16): lane holds K {h*16..h*16+15}
      ABfrag afr;
      afr.h[0] = *(const v8h*)(ap);
      afr.h[1] = *(const v8h*)(ap + 8);
      ap += 32;

      // preload all 4 B fragments, then 4 back-to-back WMMAs
      ABfrag bfr[4];
#pragma unroll
      for (int nt = 0; nt < 4; nt++) {
        const _Float16* bp = rd_cur + nt*512;   // nt*16 rows * 32 halves
        bfr[nt].h[0] = *(const v8h*)(bp);
        bfr[nt].h[1] = *(const v8h*)(bp + 8);
      }
#pragma unroll
      for (int nt = 0; nt < 4; nt++) {
        acc[nt] = __builtin_amdgcn_wmma_f32_16x16x32_f16(
            false, afr.v, false, bfr[nt].v, (short)0, acc[nt], false, false);
      }

      wait_async_done();   // next chunk resident in other buffer
      __syncthreads();     // all readers of current buffer are done

      _Float16* t;
      t = rd_cur; rd_cur = rd_nxt; rd_nxt = t;
      t = st_cur; st_cur = st_nxt; st_nxt = t;
    }
  }

  // fused BN + ReLU epilogue; C/D layout: N = Ml, M = half*8 + r
#pragma unroll
  for (int nt = 0; nt < 4; nt++) {
    int oc = (noff + nt)*16 + Ml;
    float inv = gamma[oc] * rsqrtf(var[oc] + 1e-5f);
    float mu  = mean[oc];
    float bb  = beta[oc];
    float* op = out + oc*(YD*XD) + ty*XD + tx0;
#pragma unroll
    for (int r = 0; r < 8; r++) {
      int M = half*8 + r;
      float val = (acc[nt][r] - mu)*inv + bb;
      op[M] = fmaxf(val, 0.f);
    }
  }
}

// ---------------------------------------------------------------------------
extern "C" void kernel_launch(void* const* d_in, const int* in_sizes, int n_in,
                              void* d_out, int out_size, void* d_ws, size_t ws_size,
                              hipStream_t stream) {
  const float* feats = (const float*)d_in[0];
  const float* intr  = (const float*)d_in[1];
  const float* extr  = (const float*)d_in[2];
  const float* view  = (const float*)d_in[3];
  const float* convw = (const float*)d_in[4];
  const float* gam   = (const float*)d_in[5];
  const float* bet   = (const float*)d_in[6];
  const float* mea   = (const float*)d_in[7];
  const float* var   = (const float*)d_in[8];
  float* out = (float*)d_out;

  char* ws = (char*)d_ws;
  float*     mats   = (float*)ws;                               // 384 B
  size_t off = 512;
  float*     featsT = (float*)(ws + off);                       // 5,160,960 B
  off += (size_t)NCAM*HF*WF*CCH*sizeof(float);
  _Float16*  bev    = (_Float16*)(ws + off);                    // 26,615,808 B
  off += (size_t)PADW*PADW*CBEV*sizeof(_Float16);
  _Float16*  wt     = (_Float16*)(ws + off);                    // 2,359,296 B

  setup_mats_kernel<<<1, 1, 0, stream>>>(intr, extr, view, mats);

  int tf = NCAM*CCH*HF*WF;
  transpose_feats_kernel<<<(tf + 255)/256, 256, 0, stream>>>(feats, featsT);

  int nz = PADW*PADW*CBEV/2;   // dwords
  zero_u32_kernel<<<(nz + 255)/256, 256, 0, stream>>>((unsigned*)bev, nz);

  splat_kernel<<<ZD*YD*XD, CCH, 0, stream>>>(featsT, mats, bev);

  int nw = NKCHUNK*CCH*32;
  pack_w_kernel<<<(nw + 255)/256, 256, 0, stream>>>(convw, wt);

  conv_gemm_kernel<<<196, 256, 0, stream>>>(bev, wt, gam, bet, mea, var, out);
}
